// ConvLSTM_59390807769169
// MI455X (gfx1250) — matrix operand order repaired and checked by reading, not compiled, not run on hardware
//
#include <hip/hip_runtime.h>
#include <hip/hip_bf16.h>

// ---------------------------------------------------------------------------
// ConvLSTM (2 layers, L=12, B=16, 64x64) as 9-way shift-decomposed GEMM
// (K = input channels, one WMMA K-slab = 32 channels) + fused LSTM gates.
// Matrix core: v_wmma_f32_16x16x32_bf16 (wave32, gfx1250).
// ---------------------------------------------------------------------------

typedef __attribute__((ext_vector_type(16))) __bf16 v16bf;
typedef __attribute__((ext_vector_type(8)))  float  v8f;

union Frag32 {
    unsigned int u[8];
    uint4 q[2];
    v16bf v;
};

__device__ __forceinline__ float fsig(float x) {
    return 1.0f / (1.0f + __expf(-x));
}

__device__ __forceinline__ unsigned short f2bf_bits(float f) {
    __hip_bfloat16 h = __float2bfloat16(f);
    unsigned short u;
    __builtin_memcpy(&u, &h, sizeof(u));
    return u;
}

// ---------------------------------------------------------------------------
// Zero the recurrent state region (deterministic per launch).
// ---------------------------------------------------------------------------
__global__ void zero_ws_kernel(unsigned int* __restrict__ p, unsigned long long n) {
    unsigned long long i  = (unsigned long long)blockIdx.x * blockDim.x + threadIdx.x;
    unsigned long long st = (unsigned long long)gridDim.x * blockDim.x;
    for (; i < n; i += st) p[i] = 0u;
}

// ---------------------------------------------------------------------------
// Weights: f32 OIHW -> bf16 [kyx][gate][n][c], c contiguous, zero padded to
// hid_pad rows and Cpad channels.  out[((kyx*4+g)*hid_pad + n)*Cpad + c]
// ---------------------------------------------------------------------------
__global__ void prep_w_kernel(const float* __restrict__ W, __hip_bfloat16* __restrict__ Wt,
                              int hid, int hid_pad, int cin_total, int Cpad) {
    int idx   = blockIdx.x * 256 + threadIdx.x;
    int total = 9 * 4 * hid_pad * Cpad;
    if (idx >= total) return;
    int row  = idx / Cpad;          // (kyx*4+g)*hid_pad + n
    int c    = idx - row * Cpad;
    int kyxg = row / hid_pad;       // kyx*4 + g
    int n    = row - kyxg * hid_pad;
    int kyx  = kyxg >> 2;
    int g    = kyxg & 3;
    float v = 0.0f;
    if (n < hid && c < cin_total)
        v = W[((size_t)(g * hid + n) * cin_total + c) * 9 + kyx];
    Wt[idx] = __float2bfloat16(v);
}

// ---------------------------------------------------------------------------
// One ConvLSTM timestep for one layer (geometry baked in at compile time).
//   Tile: 128 pixels (2 image rows) x 16 hidden channels x 4 gates.
//   256 threads = 8 waves; wave w owns pixels [m0+16w, m0+16w+16).
//   Raw LDS slab: 32 channels x 4 rows x 66 cols (x-halo cols stay zero),
//   channel-minor (stride 36) so A K-pairs are contiguous b32 ds loads.
//   One 64-bit weight pointer per chunk; all 72 B loads use constant IOFFSET.
// ---------------------------------------------------------------------------
#define RAW_CSTR 36
#define RAW_XW   66
#define RAW_U16  (4 * RAW_XW * RAW_CSTR)   // 9504

template <int HID, int HID_PAD, int CIN_X, int CIN_TOTAL, bool XF32>
__global__ __launch_bounds__(256) void convlstm_step_kernel(
    const float*          __restrict__ xf32,      // layer0 x source (history slice)
    const __hip_bfloat16* __restrict__ xbf,       // layer1 x source (h0 of this step)
    const __hip_bfloat16* __restrict__ hprev,     // recurrent h_{t-1} (bf16)
    const __hip_bfloat16* __restrict__ wt,        // [9*4*HID_PAD][CPAD] bf16 weights
    const float*          __restrict__ bias,      // (4*HID,) f32
    float*                __restrict__ cstate,    // (B,HID,64,64) f32, in-place
    __hip_bfloat16*       __restrict__ hnext,     // (B,HID,64,64) bf16 out
    float*                __restrict__ hout_f32,  // optional f32 h out (layer1 -> d_out)
    int xbstride)
{
    constexpr int CPAD  = 160;           // 5 chunks of 32 channels
    constexpr int NCH   = 5;
    constexpr int CPAD2 = CPAD / 2;      // u32 units per weight row

    __shared__ __align__(16) unsigned short raw[RAW_U16];

    const int tid  = threadIdx.x;
    const int lane = tid & 31;
    const int wave = tid >> 5;
    const int m0   = blockIdx.x * 128;   // first pixel of tile (same batch, 2 rows)
    const int n0   = blockIdx.y * 16;    // first hidden channel of tile
    const int half = lane >> 4;
    const int ncol = lane & 15;

    const int tb = m0 >> 12;             // batch of this tile
    const int y0 = (m0 & 4095) >> 6;     // first image row of this tile

    // zero the raw slab once: x-halo columns (never overwritten) must be 0
    {
        unsigned int* rz = reinterpret_cast<unsigned int*>(raw);
        for (int i = tid; i < RAW_U16 / 2; i += 256) rz[i] = 0u;
    }

    // staging assignment: 2 threads per (channel, row), 32 cols each
    const int sc  = tid >> 3;            // channel-in-chunk 0..31
    const int srr = (tid >> 1) & 3;      // raw row 0..3  (image row y0-1+srr)
    const int sxh = (tid & 1) << 5;      // 0 or 32

    // wave-fragment coordinates
    const int ai  = (wave << 4) + ncol;  // pixel index in tile, 0..127
    const int ax  = ai & 63;             // x
    const int atr = ai >> 6;             // tile row 0/1 (uniform per wave)

    v8f accI = {}, accF = {}, accO = {}, accG = {};

    const unsigned int* wu = reinterpret_cast<const unsigned int*>(wt);
    const unsigned wl = (unsigned)(n0 + ncol) * CPAD2 + (half << 3); // lane's u32 offset

    for (int cc = 0; cc < NCH; ++cc) {
        __syncthreads();   // previous chunk fully consumed (and zero-fill done)

        // ---- stage raw slab: 32 channels x 4 rows x 64 cols, vectorized ----
        {
            const int gc = (cc << 5) + sc;     // channel in concat([x, h])
            const int yy = y0 - 1 + srr;       // image row (may be halo)
            unsigned short vals[32];
            const bool valid = (gc < CIN_TOTAL) && ((unsigned)yy < 64u);
            if (valid) {
                if (gc < CIN_X) {
                    if (XF32) {  // layer 0: f32 history -> bf16
                        const float4* p = reinterpret_cast<const float4*>(xf32);
                        unsigned fi = ((unsigned)tb * (unsigned)xbstride +
                                       (unsigned)gc * 4096u + (unsigned)(yy * 64) + sxh) >> 2;
                        #pragma unroll
                        for (int q = 0; q < 8; ++q) {
                            float4 f = p[fi + q];
                            vals[q * 4 + 0] = f2bf_bits(f.x);
                            vals[q * 4 + 1] = f2bf_bits(f.y);
                            vals[q * 4 + 2] = f2bf_bits(f.z);
                            vals[q * 4 + 3] = f2bf_bits(f.w);
                        }
                    } else {     // layer 1: bf16 h0
                        const uint4* p = reinterpret_cast<const uint4*>(xbf);
                        unsigned ui = ((unsigned)tb * (unsigned)xbstride +
                                       (unsigned)gc * 4096u + (unsigned)(yy * 64) + sxh) >> 3;
                        #pragma unroll
                        for (int q = 0; q < 4; ++q) {
                            uint4 u = p[ui + q];
                            vals[q * 8 + 0] = (unsigned short)(u.x);
                            vals[q * 8 + 1] = (unsigned short)(u.x >> 16);
                            vals[q * 8 + 2] = (unsigned short)(u.y);
                            vals[q * 8 + 3] = (unsigned short)(u.y >> 16);
                            vals[q * 8 + 4] = (unsigned short)(u.z);
                            vals[q * 8 + 5] = (unsigned short)(u.z >> 16);
                            vals[q * 8 + 6] = (unsigned short)(u.w);
                            vals[q * 8 + 7] = (unsigned short)(u.w >> 16);
                        }
                    }
                } else {         // recurrent h_{t-1}, bf16
                    const uint4* p = reinterpret_cast<const uint4*>(hprev);
                    unsigned ui = (((unsigned)tb * HID + (unsigned)(gc - CIN_X)) * 4096u +
                                   (unsigned)(yy * 64) + sxh) >> 3;
                    #pragma unroll
                    for (int q = 0; q < 4; ++q) {
                        uint4 u = p[ui + q];
                        vals[q * 8 + 0] = (unsigned short)(u.x);
                        vals[q * 8 + 1] = (unsigned short)(u.x >> 16);
                        vals[q * 8 + 2] = (unsigned short)(u.y);
                        vals[q * 8 + 3] = (unsigned short)(u.y >> 16);
                        vals[q * 8 + 4] = (unsigned short)(u.z);
                        vals[q * 8 + 5] = (unsigned short)(u.z >> 16);
                        vals[q * 8 + 6] = (unsigned short)(u.w);
                        vals[q * 8 + 7] = (unsigned short)(u.w >> 16);
                    }
                }
            } else {
                #pragma unroll
                for (int j = 0; j < 32; ++j) vals[j] = 0;
            }
            // transpose into channel-minor slab; cols shifted by +1 (x halo)
            #pragma unroll
            for (int j = 0; j < 32; ++j)
                raw[(srr * RAW_XW + 1 + sxh + j) * RAW_CSTR + sc] = vals[j];
        }
        __syncthreads();

        if (cc + 1 < NCH) {  // prefetch next weight K-slab (global_prefetch_b8)
            __builtin_prefetch(wu + wl + (unsigned)(cc + 1) * 16u, 0, 3);
        }

        // ONE 64-bit weight pointer per chunk; every row below is a constant
        // uint4 offset that folds into the global_load IOFFSET immediate.
        const uint4* wp = reinterpret_cast<const uint4*>(wu + (wl + (unsigned)cc * 16u));

        // ---- 9 shifted WMMA steps over this 32-channel slab ----
        #pragma unroll
        for (int ky = 0; ky < 3; ++ky) {
            #pragma unroll
            for (int kx = 0; kx < 3; ++kx) {
                // A fragment: 16x32 bf16, branch-free (halo pre-zeroed in LDS)
                Frag32 a;
                const unsigned short* ap =
                    &raw[((atr + ky) * RAW_XW + ax + kx) * RAW_CSTR];
                #pragma unroll
                for (int v = 0; v < 4; ++v) {
                    a.u[v]     = *reinterpret_cast<const unsigned int*>(ap + (half << 4) + (v << 1));
                    a.u[4 + v] = *reinterpret_cast<const unsigned int*>(ap + 8 + (half << 4) + (v << 1));
                }

                // B fragments: 4 gates at compile-time uint4 offsets
                const int kyxg = (ky * 3 + kx) << 2;
                constexpr int ROWQ = HID_PAD * CPAD2 / 4;  // uint4s per (kyx,gate) block of HID_PAD rows
                Frag32 bI, bF, bO, bG;
                bI.q[0] = wp[(kyxg + 0) * ROWQ];
                bI.q[1] = wp[(kyxg + 0) * ROWQ + 1];
                bF.q[0] = wp[(kyxg + 1) * ROWQ];
                bF.q[1] = wp[(kyxg + 1) * ROWQ + 1];
                bO.q[0] = wp[(kyxg + 2) * ROWQ];
                bO.q[1] = wp[(kyxg + 2) * ROWQ + 1];
                bG.q[0] = wp[(kyxg + 3) * ROWQ];
                bG.q[1] = wp[(kyxg + 3) * ROWQ + 1];

                accI = __builtin_amdgcn_wmma_f32_16x16x32_bf16(false, a.v, false, bI.v, (short)0, accI, false, false);
                accF = __builtin_amdgcn_wmma_f32_16x16x32_bf16(false, a.v, false, bF.v, (short)0, accF, false, false);
                accO = __builtin_amdgcn_wmma_f32_16x16x32_bf16(false, a.v, false, bO.v, (short)0, accO, false, false);
                accG = __builtin_amdgcn_wmma_f32_16x16x32_bf16(false, a.v, false, bG.v, (short)0, accG, false, false);
            }
        }
    }

    // ---- fused LSTM gate epilogue ----
    // D layout: VGPR r, lane l -> M = r + 8*(l>=16), N = l&15
    const int n = n0 + ncol;
    if (n < HID) {
        const float bi = bias[n];
        const float bf = bias[HID + n];
        const float bo = bias[2 * HID + n];
        const float bg = bias[3 * HID + n];
        #pragma unroll
        for (int r = 0; r < 8; ++r) {
            int i   = (wave << 4) + (half << 3) + r;
            int m   = m0 + i;
            int pix = m & 4095;
            unsigned cidx = ((unsigned)tb * HID + (unsigned)n) * 4096u + (unsigned)pix;

            float zi = accI[r] + bi;
            float zf = accF[r] + bf;
            float zo = accO[r] + bo;
            float zg = accG[r] + bg;

            float cprev = cstate[cidx];
            float cn = fsig(zf) * cprev + fsig(zi) * tanhf(zg);
            float hn = fsig(zo) * tanhf(cn);

            cstate[cidx] = cn;
            hnext[cidx]  = __float2bfloat16(hn);
            if (hout_f32) hout_f32[cidx] = hn;
        }
    }
}

// ---------------------------------------------------------------------------
// Host launch sequence (graph-capture safe: fixed 27 launches, stream only).
// ---------------------------------------------------------------------------
extern "C" void kernel_launch(void* const* d_in, const int* in_sizes, int n_in,
                              void* d_out, int out_size, void* d_ws, size_t ws_size,
                              hipStream_t stream)
{
    (void)in_sizes; (void)n_in; (void)out_size; (void)ws_size;

    const float* hist = (const float*)d_in[0];   // (B,L,4096,1) f32
    const float* W0   = (const float*)d_in[2];   // (512,129,3,3)
    const float* b0   = (const float*)d_in[3];   // (512,)
    const float* W1   = (const float*)d_in[4];   // (48,140,3,3)
    const float* b1   = (const float*)d_in[5];   // (48,)

    const int B = 16, L = 12, HW = 4096;
    const int HID0 = 128, HID1 = 12, HID1P = 16;
    const int CIN0 = 129, CIN1 = 140;
    const int CPAD = 160;

    // workspace carve-up
    float* c0 = (float*)d_ws;                                   // B*HID0*HW f32
    float* c1 = c0 + (size_t)B * HID0 * HW;                     // B*HID1*HW f32
    __hip_bfloat16* h0a = (__hip_bfloat16*)(c1 + (size_t)B * HID1 * HW);
    __hip_bfloat16* h0b = h0a + (size_t)B * HID0 * HW;
    __hip_bfloat16* h1a = h0b + (size_t)B * HID0 * HW;
    __hip_bfloat16* h1b = h1a + (size_t)B * HID1 * HW;
    __hip_bfloat16* wt0 = h1b + (size_t)B * HID1 * HW;          // 9*4*HID0*CPAD bf16
    __hip_bfloat16* wt1 = wt0 + (size_t)9 * 4 * HID0 * CPAD;    // 9*4*HID1P*CPAD bf16

    // zero recurrent state (c0,c1,h0a/b,h1a/b) each launch
    unsigned long long state_bytes =
        (unsigned long long)B * HID0 * HW * 4 + (unsigned long long)B * HID1 * HW * 4 +
        2ull * B * HID0 * HW * 2 + 2ull * B * HID1 * HW * 2;
    zero_ws_kernel<<<2048, 256, 0, stream>>>((unsigned int*)d_ws, state_bytes / 4ull);

    prep_w_kernel<<<(9 * 4 * HID0 * CPAD + 255) / 256, 256, 0, stream>>>(
        W0, wt0, HID0, HID0, CIN0, CPAD);
    prep_w_kernel<<<(9 * 4 * HID1P * CPAD + 255) / 256, 256, 0, stream>>>(
        W1, wt1, HID1, HID1P, CIN1, CPAD);

    for (int t = 0; t < L; ++t) {
        __hip_bfloat16* h0p = (t & 1) ? h0b : h0a;
        __hip_bfloat16* h0n = (t & 1) ? h0a : h0b;
        __hip_bfloat16* h1p = (t & 1) ? h1b : h1a;
        __hip_bfloat16* h1n = (t & 1) ? h1a : h1b;

        // layer 0: x = history[:, t], 129 -> 4x128 channels
        convlstm_step_kernel<128, 128, 1, 129, true><<<dim3(512, 8), 256, 0, stream>>>(
            hist + (size_t)t * HW, nullptr, h0p, wt0, b0, c0, h0n, nullptr,
            /*xbstride=*/L * HW);

        // layer 1: x = h0 of this step, 140 -> 4x12; f32 h straight to d_out
        convlstm_step_kernel<12, 16, 128, 140, false><<<dim3(512, 1), 256, 0, stream>>>(
            nullptr, h0n, h1p, wt1, b1, c1, h1n, (float*)d_out,
            /*xbstride=*/HID0 * HW);
    }
}
